// TransformerBlock_26121991094971
// MI455X (gfx1250) — compile-verified, matrix operand order checked by reading, not run on hardware
//
#include <hip/hip_runtime.h>
#include <hip/hip_bf16.h>

// CDNA5 / gfx1250 fused transformer block.
//  - all matmuls on v_wmma_f32_16x16x32_f16
//  - flash-attention softmax (2048x2048 scores never hit HBM)
//  - K / V^T tiles double-buffered via global_load_async_to_lds_b128
//    (ASYNCcnt) so DMA overlaps WMMA; V is produced pre-transposed by the
//    QKV kernel so no per-tile transpose is needed.

typedef __attribute__((ext_vector_type(16))) _Float16 v16h;
typedef __attribute__((ext_vector_type(8)))  _Float16 v8h;
typedef __attribute__((ext_vector_type(8)))  float    v8f;
typedef __attribute__((ext_vector_type(4)))  float    v4f;

#define DP 256      // d_points
#define DM 512      // d_model
#define BB 8
#define NN 2048

// ---------------------------------------------------------------------------
// Fragment loader (LDS f16, row-major, ld multiple of 8).  CDNA5 16-bit
// A-matrix 16x32 layout: lane&15 = row, lane>>4 = K-half; element t maps to
// K = (t<8 ? t : t+8) + half*8 -> two contiguous 16B chunks per lane.  The B
// operand uses identical striping with lane&15 = column, so loading B^T
// (N-major) with this loader yields a correct B fragment.
// ---------------------------------------------------------------------------
__device__ __forceinline__ v16h ld_frag(const _Float16* base, int ld) {
  const int lane = threadIdx.x & 31;
  const int r = lane & 15;
  const int h = lane >> 4;
  const _Float16* p = base + r * ld + h * 8;
  v8h lo = *(const v8h*)(p);
  v8h hi = *(const v8h*)(p + 16);
  v16h out;
#pragma unroll
  for (int i = 0; i < 8; ++i) { out[i] = lo[i]; out[i + 8] = hi[i]; }
  return out;
}

__device__ __forceinline__ v8f wmma_f16(v16h a, v16h b, v8f c) {
  return __builtin_amdgcn_wmma_f32_16x16x32_f16(false, a, false, b, (short)0, c,
                                                false, false);
}

// Async DMA: 16 bytes global -> LDS, tracked by ASYNCcnt (ISA 08, GVS mode:
// mem = SADDR64 + VADDR32; lds = VDST32).  Generic shared-pointer low 32 bits
// give the LDS byte offset.
__device__ __forceinline__ uint32_t lds_off(const void* p) {
  return (uint32_t)(uintptr_t)p;
}
__device__ __forceinline__ void async_ld16(uint32_t lds, uint32_t goff,
                                           const void* base) {
  asm volatile("global_load_async_to_lds_b128 %0, %1, %2"
               :: "v"(lds), "v"(goff), "s"(base) : "memory");
}
__device__ __forceinline__ void wait_async0() {
  asm volatile("s_wait_asynccnt 0" ::: "memory");
}

// ---------------------------------------------------------------------------
// Kernel 1: QKV projection.  X[16384,256](f32) x W[256,512](f32) -> f16.
// Q,K stored row-major [B*N, DM];  V stored TRANSPOSED [B, DM, N] so the
// attention kernel can DMA V^T tiles with contiguous rows.
// Grid: (rows/64, 3*512/64).
// ---------------------------------------------------------------------------
__global__ __launch_bounds__(256) void qkv_kernel(
    const float* __restrict__ X, const float* __restrict__ Wq,
    const float* __restrict__ Wk, const float* __restrict__ Wv,
    _Float16* __restrict__ Q, _Float16* __restrict__ K,
    _Float16* __restrict__ Vt) {
  constexpr int LDA = DP + 8;  // 264
  __shared__ _Float16 Xs[64 * LDA];
  __shared__ _Float16 Wt[64 * LDA];  // W^T tile: Wt[n][k]

  const int tid = threadIdx.x;
  const int row0 = blockIdx.x * 64;
  const int ncomb = blockIdx.y * 64;  // 0..1535
  const int widx = ncomb / DM;
  const int n0 = ncomb % DM;
  const float* W = (widx == 0) ? Wq : (widx == 1) ? Wk : Wv;

  // Stage X tile 64x256 (f32 -> f16).
  for (int i = tid; i < 64 * (DP / 4); i += 256) {
    int r = i / (DP / 4);
    int c = (i % (DP / 4)) * 4;
    v4f x = *(const v4f*)(X + (size_t)(row0 + r) * DP + c);
    _Float16* d = &Xs[r * LDA + c];
    d[0] = (_Float16)x[0]; d[1] = (_Float16)x[1];
    d[2] = (_Float16)x[2]; d[3] = (_Float16)x[3];
  }
  // Stage W^T tile: Wt[n][k] = W[k][n0+n].
  for (int i = tid; i < DP * 64; i += 256) {
    int k = i >> 6;
    int n = i & 63;
    Wt[n * LDA + k] = (_Float16)W[(size_t)k * DM + n0 + n];
  }
  __syncthreads();

  const int wid = tid >> 5, lane = tid & 31;
  const int mw = wid >> 1;  // 0..3 : 16-row group
  const int nw = wid & 1;   // 0..1 : 32-col group
  v8f acc0 = {}, acc1 = {};
#pragma unroll
  for (int k0 = 0; k0 < DP; k0 += 32) {
    v16h a  = ld_frag(&Xs[mw * 16 * LDA + k0], LDA);
    v16h b0 = ld_frag(&Wt[(nw * 32) * LDA + k0], LDA);
    v16h b1 = ld_frag(&Wt[(nw * 32 + 16) * LDA + k0], LDA);
    acc0 = wmma_f16(a, b0, acc0);
    acc1 = wmma_f16(a, b1, acc1);
  }
  const int nn = lane & 15, mh = (lane >> 4) * 8;
  if (widx < 2) {
    _Float16* Out = (widx == 0) ? Q : K;
#pragma unroll
    for (int r = 0; r < 8; ++r) {
      int m = mw * 16 + mh + r;
      size_t base = (size_t)(row0 + m) * DM + n0 + nw * 32;
      Out[base + nn]      = (_Float16)acc0[r];
      Out[base + 16 + nn] = (_Float16)acc1[r];
    }
  } else {
    // V: transpose the 64x64 tile through LDS, then coalesced global store.
    constexpr int LDT = 72;
    _Float16* Tt = Xs;  // reuse (64*72 <= 64*264)
    __syncthreads();    // Xs/Wt reads all done
#pragma unroll
    for (int r = 0; r < 8; ++r) {
      int m = mw * 16 + mh + r;
      Tt[m * LDT + nw * 32 + nn]      = (_Float16)acc0[r];
      Tt[m * LDT + nw * 32 + 16 + nn] = (_Float16)acc1[r];
    }
    __syncthreads();
    const int bb = row0 / NN;
    const int key0 = row0 & (NN - 1);
    for (int i = tid; i < 64 * 64; i += 256) {
      int d = i >> 6;    // dim within tile
      int kk = i & 63;   // key within tile
      Vt[((size_t)bb * DM + n0 + d) * NN + key0 + kk] = Tt[kk * LDT + d];
    }
  }
}

// ---------------------------------------------------------------------------
// Kernel 2: flash attention.  Block = (batch b, 64 query rows); 8 waves:
// mw = wid>>1 (16-row group), nw = wid&1.  O accumulator 16x256 f32/wave in
// registers.  K and V^T tiles (32 keys) double-buffered, filled by async DMA.
// ---------------------------------------------------------------------------
#define LDQ  (DM + 8)   // 520 halfs (1040 B, 16B aligned rows)
#define LDV  40         // 32 keys + 8 pad (80 B rows, 16B aligned)
#define LDSS 36
#define NTILES (NN / 32)

__global__ __launch_bounds__(256) void attn_kernel(
    const _Float16* __restrict__ Q, const _Float16* __restrict__ K,
    const _Float16* __restrict__ Vt, _Float16* __restrict__ O) {
  __shared__ _Float16 Qs[64 * LDQ];       // 66.5 KB
  __shared__ _Float16 Ks[2][32 * LDQ];    // 2 x 33.3 KB
  __shared__ _Float16 Vs[2][DM * LDV];    // 2 x 41 KB   V^T: Vs[d][key]
  __shared__ float    Sraw[64 * LDSS];
  __shared__ _Float16 Ps[64 * LDV];
  __shared__ float m_sh[64], l_sh[64], a_sh[64];

  const int tid = threadIdx.x;
  const int b = blockIdx.y;
  const int row0 = blockIdx.x * 64;
  const size_t qbase = ((size_t)b * NN + row0) * DM;
  const float scale = 0.044194173824159216f;  // 1/sqrt(512)

  // Stage Q tile 64x512 once (synchronous vector copies).
  for (int i = tid; i < 64 * (DM / 8); i += 256) {
    int r = i / (DM / 8);
    int c = (i % (DM / 8)) * 8;
    *(v8h*)&Qs[r * LDQ + c] = *(const v8h*)(Q + qbase + (size_t)r * DM + c);
  }
  if (tid < 64) { m_sh[tid] = -1e30f; l_sh[tid] = 0.0f; }

  const int wid = tid >> 5, lane = tid & 31;
  const int mw = wid >> 1, nw = wid & 1;
  const int nn = lane & 15, mh = (lane >> 4) * 8;
  v8f acc[16];
#pragma unroll
  for (int i = 0; i < 16; ++i) acc[i] = (v8f){};

  // Async-DMA one 32-key tile (K: 32x512 halfs, V^T: 512x32 halfs) into buf.
  const uint32_t kls[2] = {lds_off(&Ks[0][0]), lds_off(&Ks[1][0])};
  const uint32_t vls[2] = {lds_off(&Vs[0][0]), lds_off(&Vs[1][0])};
  const uint32_t kbase0 = (uint32_t)((size_t)b * NN * DM * 2);
  const uint32_t vbase0 = (uint32_t)((size_t)b * DM * NN * 2);
  auto stage = [&](int kt, int buf) {
    const uint32_t kb = kbase0 + (uint32_t)kt * 32u * DM * 2u;  // tile rows
    const uint32_t vb = vbase0 + (uint32_t)kt * 32u * 2u;       // key offset
#pragma unroll
    for (int j = 0; j < 8; ++j) {            // K: 2048 x 16B chunks
      int c = tid + j * 256;
      int r = c >> 6, cc = c & 63;
      async_ld16(kls[buf] + (uint32_t)(r * (LDQ * 2) + cc * 16),
                 kb + (uint32_t)(r * (DM * 2) + cc * 16), K);
    }
#pragma unroll
    for (int j = 0; j < 8; ++j) {            // V^T: 512 rows x 4 chunks
      int c = tid + j * 256;
      int d = c >> 2, cc = c & 3;
      async_ld16(vls[buf] + (uint32_t)(d * (LDV * 2) + cc * 16),
                 vb + (uint32_t)(d * (NN * 2) + cc * 16), Vt);
    }
  };

  stage(0, 0);
  wait_async0();
  __syncthreads();

  int buf = 0;
  for (int kt = 0; kt < NTILES; ++kt) {
    if (kt + 1 < NTILES) stage(kt + 1, buf ^ 1);  // prefetch next tile

    // S = (Q K^T) * scale : one 16x16 block per wave.
    v8f s = (v8f){};
#pragma unroll
    for (int k0 = 0; k0 < DM; k0 += 32) {
      v16h a   = ld_frag(&Qs[mw * 16 * LDQ + k0], LDQ);
      v16h bfr = ld_frag(&Ks[buf][nw * 16 * LDQ + k0], LDQ);
      s = wmma_f16(a, bfr, s);
    }
    {
      float* dst = &Sraw[(mw * 16 + mh) * LDSS + nw * 16 + nn];
#pragma unroll
      for (int r = 0; r < 8; ++r) dst[r * LDSS] = s[r] * scale;
    }
    __syncthreads();

    // Online softmax: one thread per query row.
    if (tid < 64) {
      float mold = m_sh[tid], mnew = mold;
      float* srow = &Sraw[tid * LDSS];
#pragma unroll
      for (int c = 0; c < 32; ++c) mnew = fmaxf(mnew, srow[c]);
      float alpha = __expf(mold - mnew);
      float lsum = 0.0f;
      _Float16* prow = &Ps[tid * LDV];
#pragma unroll
      for (int c = 0; c < 32; ++c) {
        float p = __expf(srow[c] - mnew);
        prow[c] = (_Float16)p;
        lsum += p;
      }
      m_sh[tid] = mnew;
      l_sh[tid] = l_sh[tid] * alpha + lsum;
      a_sh[tid] = alpha;
    }
    __syncthreads();

    // Rescale running O, accumulate P @ V over 16 dim-blocks.
    float ar[8];
#pragma unroll
    for (int r = 0; r < 8; ++r) ar[r] = a_sh[mw * 16 + mh + r];
    v16h p = ld_frag(&Ps[mw * 16 * LDV], LDV);
#pragma unroll
    for (int cb = 0; cb < 16; ++cb) {
#pragma unroll
      for (int r = 0; r < 8; ++r) acc[cb][r] *= ar[r];
      v16h bfr = ld_frag(&Vs[buf][(nw * 256 + cb * 16) * LDV], LDV);
      acc[cb] = wmma_f16(p, bfr, acc[cb]);
    }

    wait_async0();      // own DMA for tile kt+1 done
    __syncthreads();    // publish next buffer / retire this one
    buf ^= 1;
  }

  // Finalize: O /= l, store f16 row-major.
  float inv[8];
#pragma unroll
  for (int r = 0; r < 8; ++r) inv[r] = 1.0f / l_sh[mw * 16 + mh + r];
#pragma unroll
  for (int cb = 0; cb < 16; ++cb) {
    int n = nw * 256 + cb * 16 + nn;
#pragma unroll
    for (int r = 0; r < 8; ++r) {
      int m = mw * 16 + mh + r;
      O[qbase + (size_t)m * DM + n] = (_Float16)(acc[cb][r] * inv[r]);
    }
  }
}

// ---------------------------------------------------------------------------
// Kernel 3: fc2 + bias + leaky_relu(0.2) + residual.  [16384,512]x[512,256].
// ---------------------------------------------------------------------------
__global__ __launch_bounds__(256) void fc2_kernel(
    const _Float16* __restrict__ A, const float* __restrict__ W2,
    const float* __restrict__ b2, const float* __restrict__ X,
    float* __restrict__ Out) {
  constexpr int LDA = DM + 8;  // 520
  __shared__ _Float16 As[64 * LDA];
  __shared__ _Float16 Wt[64 * LDA];
  const int tid = threadIdx.x;
  const int row0 = blockIdx.x * 64;
  const int n0 = blockIdx.y * 64;

  for (int i = tid; i < 64 * (DM / 8); i += 256) {
    int r = i / (DM / 8);
    int c = (i % (DM / 8)) * 8;
    *(v8h*)&As[r * LDA + c] = *(const v8h*)(A + (size_t)(row0 + r) * DM + c);
  }
  for (int i = tid; i < DM * 64; i += 256) {
    int k = i >> 6;
    int n = i & 63;
    Wt[n * LDA + k] = (_Float16)W2[(size_t)k * DP + n0 + n];
  }
  __syncthreads();

  const int wid = tid >> 5, lane = tid & 31;
  const int mw = wid >> 1, nw = wid & 1;
  v8f acc0 = {}, acc1 = {};
#pragma unroll
  for (int k0 = 0; k0 < DM; k0 += 32) {
    v16h a  = ld_frag(&As[mw * 16 * LDA + k0], LDA);
    v16h b0 = ld_frag(&Wt[(nw * 32) * LDA + k0], LDA);
    v16h b1 = ld_frag(&Wt[(nw * 32 + 16) * LDA + k0], LDA);
    acc0 = wmma_f16(a, b0, acc0);
    acc1 = wmma_f16(a, b1, acc1);
  }
  const int nn = lane & 15, mh = (lane >> 4) * 8;
#pragma unroll
  for (int r = 0; r < 8; ++r) {
    int m = mw * 16 + mh + r;
    size_t orow = (size_t)(row0 + m) * DP;
    int n = n0 + nw * 32 + nn;
    float v0 = acc0[r] + b2[n];
    v0 = (v0 > 0.0f) ? v0 : 0.2f * v0;
    Out[orow + n] = v0 + X[orow + n];
    float v1 = acc1[r] + b2[n + 16];
    v1 = (v1 > 0.0f) ? v1 : 0.2f * v1;
    Out[orow + n + 16] = v1 + X[orow + n + 16];
  }
}

// ---------------------------------------------------------------------------
extern "C" void kernel_launch(void* const* d_in, const int* in_sizes, int n_in,
                              void* d_out, int out_size, void* d_ws,
                              size_t ws_size, hipStream_t stream) {
  (void)in_sizes; (void)n_in; (void)out_size; (void)ws_size;
  const float* features = (const float*)d_in[0];
  const float* Wq = (const float*)d_in[1];
  const float* Wk = (const float*)d_in[2];
  const float* Wv = (const float*)d_in[3];
  const float* W2 = (const float*)d_in[4];
  const float* b2 = (const float*)d_in[5];
  float* out = (float*)d_out;

  const size_t rows = (size_t)BB * NN;  // 16384
  _Float16* Qg  = (_Float16*)d_ws;      // 16 MB each, 64 MB total
  _Float16* Kg  = Qg + rows * DM;
  _Float16* Vtg = Kg + rows * DM;       // [B, DM, N] transposed
  _Float16* Og  = Vtg + rows * DM;

  dim3 blk(256);
  qkv_kernel<<<dim3((unsigned)(rows / 64), (3 * DM) / 64), blk, 0, stream>>>(
      features, Wq, Wk, Wv, Qg, Kg, Vtg);
  attn_kernel<<<dim3(NN / 64, BB), blk, 0, stream>>>(Qg, Kg, Vtg, Og);
  fc2_kernel<<<dim3((unsigned)(rows / 64), DP / 64), blk, 0, stream>>>(
      Og, W2, b2, features, out);
}
